// LocalInteractionLayer_71846212927759
// MI455X (gfx1250) — matrix operand order repaired and checked by reading, not compiled, not run on hardware
//
#include <hip/hip_runtime.h>
#include <hip/hip_bf16.h>
#include <stdint.h>

#define NB   8
#define LA   512
#define LB   512
#define DD   128
#define NEGV -10000000.0f

typedef __attribute__((ext_vector_type(16))) __bf16          v16bf;
typedef __attribute__((ext_vector_type(8)))  float           v8f;
typedef __attribute__((ext_vector_type(8)))  unsigned short  u16x8;

union FragU {
    struct { u16x8 lo, hi; } p;
    v16bf v;
};

__device__ __forceinline__ unsigned short f32_to_bf16_bits(float x) {
    union { float f; uint32_t u; } c; c.f = x;
    uint32_t u = c.u;
    u += 0x7FFFu + ((u >> 16) & 1u);          // round-to-nearest-even truncation
    return (unsigned short)(u >> 16);
}
__device__ __forceinline__ float bf16_bits_to_f32(unsigned short h) {
    union { uint32_t u; float f; } c; c.u = ((uint32_t)h) << 16;
    return c.f;
}

// ---------------------------------------------------------------------------
// K0: convert fp32 [B, L, D] -> split-bf16 transposed planes [B, D, L]
// ---------------------------------------------------------------------------
__global__ __launch_bounds__(256) void k_cvt_transpose(
    const float* __restrict__ src, unsigned short* __restrict__ dst_hi,
    unsigned short* __restrict__ dst_lo, int L) {
    int idx = blockIdx.x * 256 + threadIdx.x;        // enumerates (b, d, l), l fastest
    int l  = idx % L;
    int r  = idx / L;
    int d  = r % DD;
    int b  = r / DD;
    float x = src[((size_t)b * L + l) * DD + d];
    unsigned short hi = f32_to_bf16_bits(x);
    float rem = x - bf16_bits_to_f32(hi);
    dst_hi[idx] = hi;
    dst_lo[idx] = f32_to_bf16_bits(rem);
}

// ---------------------------------------------------------------------------
// K1: fused dot + L1-cdist + mask + sigmoid -> S[b, i, j] (fp32)
//     64x64 output tile per 256-thread block, 4x4 register micro-tile,
//     K chunked by 32 through LDS (pad 33 -> conflict-free b reads).
// ---------------------------------------------------------------------------
__global__ __launch_bounds__(256) void k_attn_scores(
    const float* __restrict__ a, const float* __restrict__ b,
    const unsigned char* __restrict__ mask_a, const unsigned char* __restrict__ mask_b,
    const float* __restrict__ alpha_p, const float* __restrict__ beta_p,
    float* __restrict__ S) {
    __shared__ float aS[64][33];
    __shared__ float bS[64][33];
    const float alpha = *alpha_p;
    const float beta  = *beta_p;
    int bb = blockIdx.x;
    int i0 = blockIdx.y * 64;
    int j0 = blockIdx.z * 64;
    int tid = threadIdx.x;
    int ti = tid >> 4, tj = tid & 15;

    float dotv[4][4] = {};
    float l1v[4][4]  = {};
    const float* aBase = a + ((size_t)bb * LA + i0) * DD;
    const float* bBase = b + ((size_t)bb * LB + j0) * DD;

    for (int d0 = 0; d0 < DD; d0 += 32) {
        __syncthreads();
        #pragma unroll
        for (int t = 0; t < 8; ++t) {              // 2048 floats per matrix, 8/thread
            int flat = tid + t * 256;
            int row = flat >> 5, col = flat & 31;
            aS[row][col] = aBase[(size_t)row * DD + d0 + col];
            bS[row][col] = bBase[(size_t)row * DD + d0 + col];
        }
        __syncthreads();
        for (int d = 0; d < 32; ++d) {
            float av[4], bv[4];
            #pragma unroll
            for (int r = 0; r < 4; ++r) av[r] = aS[ti * 4 + r][d];
            #pragma unroll
            for (int c = 0; c < 4; ++c) bv[c] = bS[tj * 4 + c][d];
            #pragma unroll
            for (int r = 0; r < 4; ++r)
                #pragma unroll
                for (int c = 0; c < 4; ++c) {
                    dotv[r][c] = __builtin_fmaf(av[r], bv[c], dotv[r][c]);
                    l1v[r][c] += __builtin_fabsf(av[r] - bv[c]);
                }
        }
    }
    #pragma unroll
    for (int r = 0; r < 4; ++r)
        #pragma unroll
        for (int c = 0; c < 4; ++c) {
            int i = i0 + ti * 4 + r;
            int j = j0 + tj * 4 + c;
            bool m = mask_a[bb * LA + i] && mask_b[bb * LB + j];
            float nrm = m ? dotv[r][c] * alpha : NEGV;
            float dif = m ? -beta * l1v[r][c] : NEGV;
            float sig = 1.0f / (1.0f + __expf(-dif));   // sigmoid(NEG) underflows to 0
            S[((size_t)bb * LA + i) * LB + j] = nrm * sig;
        }
}

// ---------------------------------------------------------------------------
// K2a: row softmax -> split-bf16 P[b, i, j]
// ---------------------------------------------------------------------------
__global__ __launch_bounds__(256) void k_softmax_rows(
    const float* __restrict__ S, unsigned short* __restrict__ Phi,
    unsigned short* __restrict__ Plo) {
    __shared__ float red[256];
    int b = blockIdx.y, i = blockIdx.x, tid = threadIdx.x;
    const float* row = S + ((size_t)b * LA + i) * LB;
    float x0 = row[tid], x1 = row[tid + 256];
    red[tid] = fmaxf(x0, x1);
    __syncthreads();
    for (int s = 128; s > 0; s >>= 1) {
        if (tid < s) red[tid] = fmaxf(red[tid], red[tid + s]);
        __syncthreads();
    }
    float rmax = red[0];
    __syncthreads();
    float e0 = __expf(x0 - rmax), e1 = __expf(x1 - rmax);
    red[tid] = e0 + e1;
    __syncthreads();
    for (int s = 128; s > 0; s >>= 1) {
        if (tid < s) red[tid] += red[tid + s];
        __syncthreads();
    }
    float inv = 1.0f / red[0];
    size_t base = ((size_t)b * LA + i) * LB;
    float p0 = e0 * inv, p1 = e1 * inv;
    unsigned short h0 = f32_to_bf16_bits(p0);
    unsigned short h1 = f32_to_bf16_bits(p1);
    Phi[base + tid]       = h0;
    Plo[base + tid]       = f32_to_bf16_bits(p0 - bf16_bits_to_f32(h0));
    Phi[base + tid + 256] = h1;
    Plo[base + tid + 256] = f32_to_bf16_bits(p1 - bf16_bits_to_f32(h1));
}

// ---------------------------------------------------------------------------
// K2b: column softmax -> split-bf16 P2T[b, j, i]  (transposed for WMMA A-frags)
// ---------------------------------------------------------------------------
__global__ __launch_bounds__(256) void k_softmax_cols(
    const float* __restrict__ S, unsigned short* __restrict__ Phi,
    unsigned short* __restrict__ Plo) {
    __shared__ float red[256];
    int b = blockIdx.y, j = blockIdx.x, tid = threadIdx.x;
    const float* base = S + (size_t)b * LA * LB + j;
    float x0 = base[(size_t)tid * LB], x1 = base[(size_t)(tid + 256) * LB];
    red[tid] = fmaxf(x0, x1);
    __syncthreads();
    for (int s = 128; s > 0; s >>= 1) {
        if (tid < s) red[tid] = fmaxf(red[tid], red[tid + s]);
        __syncthreads();
    }
    float cmax = red[0];
    __syncthreads();
    float e0 = __expf(x0 - cmax), e1 = __expf(x1 - cmax);
    red[tid] = e0 + e1;
    __syncthreads();
    for (int s = 128; s > 0; s >>= 1) {
        if (tid < s) red[tid] += red[tid + s];
        __syncthreads();
    }
    float inv = 1.0f / red[0];
    size_t obase = ((size_t)b * LB + j) * LA;
    float p0 = e0 * inv, p1 = e1 * inv;
    unsigned short h0 = f32_to_bf16_bits(p0);
    unsigned short h1 = f32_to_bf16_bits(p1);
    Phi[obase + tid]       = h0;
    Plo[obase + tid]       = f32_to_bf16_bits(p0 - bf16_bits_to_f32(h0));
    Phi[obase + tid + 256] = h1;
    Plo[obase + tid + 256] = f32_to_bf16_bits(p1 - bf16_bits_to_f32(h1));
}

// ---------------------------------------------------------------------------
// K3: Out[b, M, 128] = P[b, M, K] @ V[b, K, 128] via v_wmma_f32_16x16x32_bf16
//     P row-major [M,K]; V supplied transposed [128, K] so B-fragments are
//     K-contiguous. Split-bf16 compensation with THREE independent
//     accumulators so the 3 WMMAs per K-step have no D->C dependency chain.
//     One wave per 16x16 output tile; EXEC all-ones (no divergence).
// ---------------------------------------------------------------------------
__global__ __launch_bounds__(128) void k_wmma_pv(
    const unsigned short* __restrict__ Phi, const unsigned short* __restrict__ Plo,
    const unsigned short* __restrict__ Vhi, const unsigned short* __restrict__ Vlo,
    float* __restrict__ Out, int M, int K) {
    int wid  = blockIdx.x * 4 + (threadIdx.x >> 5);
    int lane = threadIdx.x & 31;
    int tilesN = DD >> 4;                      // 8
    int perB   = (M >> 4) * tilesN;            // 256
    int b  = wid / perB;
    int t  = wid % perB;
    int m0 = (t / tilesN) * 16;
    int n0 = (t % tilesN) * 16;
    int half = lane >> 4;                      // 0: lanes 0-15, 1: lanes 16-31
    int ml   = lane & 15;

    const unsigned short* Ph = Phi + ((size_t)b * M + m0 + ml) * K;
    const unsigned short* Pl = Plo + ((size_t)b * M + m0 + ml) * K;
    const unsigned short* Vh = Vhi + ((size_t)b * DD + n0 + ml) * K;
    const unsigned short* Vl = Vlo + ((size_t)b * DD + n0 + ml) * K;

    v8f acc0 = {};   // Phi * Vhi
    v8f acc1 = {};   // Phi * Vlo
    v8f acc2 = {};   // Plo * Vhi
    for (int k0 = 0; k0 < K; k0 += 32) {
        // prefetch next K-step's rows (global_prefetch_b8; warms WGP$/L2 path)
        __builtin_prefetch(Ph + k0 + 32, 0, 1);
        __builtin_prefetch(Vh + k0 + 32, 0, 1);
        // A frag (16x32 bf16): lane<16 -> K{0..7,16..23}; lane>=16 -> K{8..15,24..31}
        int aoff = k0 + half * 8;
        FragU ah, al, bh, bl;
        ah.p.lo = *(const u16x8*)(Ph + aoff);
        ah.p.hi = *(const u16x8*)(Ph + aoff + 16);
        al.p.lo = *(const u16x8*)(Pl + aoff);
        al.p.hi = *(const u16x8*)(Pl + aoff + 16);
        // B frag (32x16 bf16): lane<16 -> K{0..7,16..23}; lane>=16 -> K{8..15,24..31}
        int boff = k0 + half * 8;
        bh.p.lo = *(const u16x8*)(Vh + boff);
        bh.p.hi = *(const u16x8*)(Vh + boff + 16);
        bl.p.lo = *(const u16x8*)(Vl + boff);
        bl.p.hi = *(const u16x8*)(Vl + boff + 16);
        // (Phi+Plo)(Vhi+Vlo) ~= Phi*Vhi + Phi*Vlo + Plo*Vhi  (drop lo*lo)
        acc0 = __builtin_amdgcn_wmma_f32_16x16x32_bf16(false, ah.v, false, bh.v,
                                                       (short)0, acc0, false, false);
        acc1 = __builtin_amdgcn_wmma_f32_16x16x32_bf16(false, ah.v, false, bl.v,
                                                       (short)0, acc1, false, false);
        acc2 = __builtin_amdgcn_wmma_f32_16x16x32_bf16(false, al.v, false, bh.v,
                                                       (short)0, acc2, false, false);
    }
    v8f acc = (acc0 + acc1) + acc2;
    // D layout: lane<16 -> N=lane, M=vgpr; lane>=16 -> N=lane-16, M=8+vgpr
    float* o = Out + ((size_t)b * M + m0 + half * 8) * DD + n0 + ml;
    #pragma unroll
    for (int v = 0; v < 8; ++v) o[(size_t)v * DD] = acc[v];
}

// ---------------------------------------------------------------------------
extern "C" void kernel_launch(void* const* d_in, const int* in_sizes, int n_in,
                              void* d_out, int out_size, void* d_ws, size_t ws_size,
                              hipStream_t stream) {
    (void)in_sizes; (void)n_in; (void)out_size; (void)ws_size;
    const float*         a      = (const float*)d_in[0];
    const float*         b      = (const float*)d_in[1];
    const float*         alphaP = (const float*)d_in[2];
    const float*         betaP  = (const float*)d_in[3];
    const unsigned char* maskA  = (const unsigned char*)d_in[4];
    const unsigned char* maskB  = (const unsigned char*)d_in[5];
    float* out = (float*)d_out;

    // workspace layout (bytes)
    char* ws = (char*)d_ws;
    const size_t szS   = (size_t)NB * LA * LB * sizeof(float);          // 8 MiB
    const size_t szP   = (size_t)NB * LA * LB * sizeof(unsigned short); // 4 MiB each
    const size_t szT   = (size_t)NB * DD * LA * sizeof(unsigned short); // 1 MiB each
    float*          S      = (float*)ws;                       ws += szS;
    unsigned short* P_hi   = (unsigned short*)ws;              ws += szP;
    unsigned short* P_lo   = (unsigned short*)ws;              ws += szP;
    unsigned short* P2T_hi = (unsigned short*)ws;              ws += szP;
    unsigned short* P2T_lo = (unsigned short*)ws;              ws += szP;
    unsigned short* aT_hi  = (unsigned short*)ws;              ws += szT;
    unsigned short* aT_lo  = (unsigned short*)ws;              ws += szT;
    unsigned short* bT_hi  = (unsigned short*)ws;              ws += szT;
    unsigned short* bT_lo  = (unsigned short*)ws;              ws += szT;

    // K0: split-bf16 transposed copies of a and b
    int nT = NB * DD * LA;                                     // 524288
    hipLaunchKernelGGL(k_cvt_transpose, dim3(nT / 256), dim3(256), 0, stream,
                       a, aT_hi, aT_lo, LA);
    hipLaunchKernelGGL(k_cvt_transpose, dim3(nT / 256), dim3(256), 0, stream,
                       b, bT_hi, bT_lo, LB);

    // K1: fused scores
    hipLaunchKernelGGL(k_attn_scores, dim3(NB, LA / 64, LB / 64), dim3(256), 0, stream,
                       a, b, maskA, maskB, alphaP, betaP, S);

    // K2: softmaxes -> split-bf16 probability matrices
    hipLaunchKernelGGL(k_softmax_rows, dim3(LA, NB), dim3(256), 0, stream,
                       S, P_hi, P_lo);
    hipLaunchKernelGGL(k_softmax_cols, dim3(LB, NB), dim3(256), 0, stream,
                       S, P2T_hi, P2T_lo);

    // K3: WMMA matmuls
    // a_mac[b,i,d] = sum_j P[i,j]  * b[j,d]   (V = bT planes)
    int blocksA = (NB * (LA / 16) * (DD / 16)) / 4;            // 512
    hipLaunchKernelGGL(k_wmma_pv, dim3(blocksA), dim3(128), 0, stream,
                       P_hi, P_lo, bT_hi, bT_lo, out, LA, LB);
    // b_mac[b,j,d] = sum_i P2[i,j] * a[i,d]   (A = P2T, V = aT planes)
    int blocksB = (NB * (LB / 16) * (DD / 16)) / 4;            // 512
    hipLaunchKernelGGL(k_wmma_pv, dim3(blocksB), dim3(128), 0, stream,
                       P2T_hi, P2T_lo, aT_hi, aT_lo,
                       out + (size_t)NB * LA * DD, LB, LA);
}
